// GCLSTMCell_12927851561077
// MI455X (gfx1250) — matrix operand order repaired and checked by reading, not compiled, not run on hardware
//
#include <hip/hip_runtime.h>
#include <hip/hip_bf16.h>
#include <math.h>

// ---------------- problem constants ----------------
#define BB     16
#define NN     20000
#define D_IN   2
#define UU     64
#define EE     320000
#define IN_SZ  66                 // D_IN + U
#define FB     (IN_SZ * BB)       // 1056 columns of x0/x1
#define KPAD   160                // 2*IN_SZ = 132 padded to 5*32
#define GCOLS  256                // 4*U
#define MROWS  (BB * NN)          // 320000 GEMM rows

typedef unsigned short u16;
typedef __attribute__((ext_vector_type(16))) __bf16 v16bf;
typedef __attribute__((ext_vector_type(8)))  __bf16 v8bf;
typedef __attribute__((ext_vector_type(8)))  float  v8f;

// float -> bf16 round-to-nearest-even
static __device__ __forceinline__ u16 f2bf(float f) {
  union { float f; unsigned u; } v; v.f = f;
  unsigned r = v.u + 0x7FFFu + ((v.u >> 16) & 1u);
  return (u16)(r >> 16);
}
static __device__ __forceinline__ float bf2f(u16 h) {
  union { unsigned u; float f; } v; v.u = ((unsigned)h) << 16;
  return v.f;
}

// ---------------- CSR build ----------------
__global__ void zero_counts_kernel(int* counts, int n) {
  int i = blockIdx.x * blockDim.x + threadIdx.x;
  if (i < n) counts[i] = 0;
}

__global__ void count_rows_kernel(const int* __restrict__ rows, int* counts, int e) {
  int i = blockIdx.x * blockDim.x + threadIdx.x;
  if (i < e) atomicAdd(&counts[rows[i]], 1);
}

__global__ __launch_bounds__(1024) void scan_kernel(const int* __restrict__ counts,
                                                    int* row_start, int* cursor, int n) {
  __shared__ int s[1024];
  __shared__ int carry_s;
  int t = threadIdx.x;
  if (t == 0) carry_s = 0;
  __syncthreads();
  for (int base = 0; base < n; base += 1024) {
    int x = (base + t < n) ? counts[base + t] : 0;
    s[t] = x;
    __syncthreads();
    #pragma unroll
    for (int off = 1; off < 1024; off <<= 1) {
      int v = (t >= off) ? s[t - off] : 0;
      __syncthreads();
      s[t] += v;
      __syncthreads();
    }
    int incl  = s[t];
    int carry = carry_s;
    if (base + t < n) {
      int excl = carry + incl - x;
      row_start[base + t] = excl;
      cursor[base + t]    = excl;
    }
    __syncthreads();
    if (t == 1023) carry_s = carry + incl;
    __syncthreads();
  }
  if (t == 0) row_start[n] = carry_s;
}

__global__ void scatter_kernel(const int* __restrict__ rows, const int* __restrict__ cols,
                               const float* __restrict__ vals, int* cursor,
                               int* col_s, float* val_s, int e) {
  int i = blockIdx.x * blockDim.x + threadIdx.x;
  if (i < e) {
    int p = atomicAdd(&cursor[rows[i]], 1);
    col_s[p] = cols[i];
    val_s[p] = vals[i];
  }
}

// ---------------- pack x0 = concat(inputs, h) into bf16 [n][f][b] ----------------
__global__ void pack_x0_kernel(const float* __restrict__ inputs, const float* __restrict__ hx,
                               u16* __restrict__ x0bf) {
  int idx = blockIdx.x * blockDim.x + threadIdx.x;   // 20000*1056 = 21,120,000
  if (idx >= NN * FB) return;
  int b = idx & 15;
  int f = (idx >> 4) % IN_SZ;
  int n = idx / FB;
  float x = (f < D_IN) ? inputs[(size_t)b * (NN * D_IN) + n * D_IN + f]
                       : hx[(size_t)b * (NN * UU) + n * UU + (f - D_IN)];
  x0bf[idx] = f2bf(x);
}

// ---------------- pack W transposed + K-padded: Wpack[col][k] ----------------
__global__ void pack_w_kernel(const float* __restrict__ W, u16* __restrict__ Wpack) {
  int idx = blockIdx.x * blockDim.x + threadIdx.x;   // 256*160
  if (idx >= GCOLS * KPAD) return;
  int col = idx / KPAD;
  int k   = idx - col * KPAD;
  u16 v = 0;
  if (k < 2 * IN_SZ) v = f2bf(W[(size_t)k * GCOLS + col]);
  Wpack[idx] = v;
}

// ---------------- SpMM: x1[n,:] = sum_e val * x0[col(e),:]  (CSR gather, f32 accum) ----
__global__ __launch_bounds__(256) void spmm_kernel(const u16* __restrict__ x0bf,
                                                   const int* __restrict__ row_start,
                                                   const int* __restrict__ col_s,
                                                   const float* __restrict__ val_s,
                                                   u16* __restrict__ x1bf) {
  int n = blockIdx.x;
  int t = threadIdx.x;
  float a0 = 0.f, a1 = 0.f, a2 = 0.f, a3 = 0.f, a4 = 0.f;
  int e0 = row_start[n], e1 = row_start[n + 1];
  for (int e = e0; e < e1; ++e) {
    float v = val_s[e];
    const u16* xr = x0bf + (size_t)col_s[e] * FB;
    a0 += v * bf2f(xr[t]);
    a1 += v * bf2f(xr[t + 256]);
    a2 += v * bf2f(xr[t + 512]);
    a3 += v * bf2f(xr[t + 768]);
    if (t < FB - 1024) a4 += v * bf2f(xr[t + 1024]);
  }
  u16* orow = x1bf + (size_t)n * FB;
  orow[t]       = f2bf(a0);
  orow[t + 256] = f2bf(a1);
  orow[t + 512] = f2bf(a2);
  orow[t + 768] = f2bf(a3);
  if (t < FB - 1024) orow[t + 1024] = f2bf(a4);
}

// ---------------- fused GEMM (bf16 WMMA) + LSTM gates ----------------
// Block: 256 threads = 8 waves; each wave owns a 16-row strip x 256 cols.
// A (16x32 bf16 frags) staged via LDS; B frags read contiguously from Wpack (L2).
__global__ __launch_bounds__(256) void gemm_lstm_kernel(const u16* __restrict__ x0bf,
                                                        const u16* __restrict__ x1bf,
                                                        const u16* __restrict__ Wpack,
                                                        const float* __restrict__ bias,
                                                        const float* __restrict__ cx,
                                                        float* __restrict__ out_h,
                                                        float* __restrict__ out_c) {
  __shared__ u16 sA[128 * KPAD];               // 40 KB
  const int tid      = threadIdx.x;
  const int rowblock = blockIdx.x * 128;       // 2500 blocks * 128 rows = 320000

  // warm L2/L0 with the weight panel this wave is about to stream
  __builtin_prefetch(Wpack + tid * 128, 0, 3);

  // stage A: interleaved x0/x1 features for 128 (b,n) rows, K padded to 160
  for (int idx = tid; idx < 128 * KPAD; idx += 256) {
    int m = idx / KPAD;
    int k = idx - m * KPAD;
    u16 v = 0;
    if (k < 2 * IN_SZ) {
      int r = rowblock + m;                    // global GEMM row = b*N + n
      int n = r % NN;
      int b = r / NN;
      size_t src = (size_t)n * FB + (k >> 1) * BB + b;
      v = (k & 1) ? x1bf[src] : x0bf[src];
    }
    sA[idx] = v;
  }
  __syncthreads();

  const int w    = tid >> 5;                   // wave id: row-tile
  const int l    = tid & 31;
  const int lmod = l & 15;
  const int lhi  = l >> 4;

  v8f acc[16];
  v8f zero = {0.f, 0.f, 0.f, 0.f, 0.f, 0.f, 0.f, 0.f};
  #pragma unroll
  for (int i = 0; i < 16; ++i) acc[i] = zero;

  #pragma unroll
  for (int ks = 0; ks < KPAD / 32; ++ks) {
    // A fragment: row = l%16,
    // K = ks*32 + (l/16)*16 + j  -> two 16B LDS reads
    const u16* ap = sA + (size_t)(w * 16 + lmod) * KPAD + ks * 32 + lhi * 16;
    v8bf alo = *reinterpret_cast<const v8bf*>(ap);
    v8bf ahi = *reinterpret_cast<const v8bf*>(ap + 8);
    v16bf a;
    #pragma unroll
    for (int j = 0; j < 8; ++j) { a[j] = alo[j]; a[j + 8] = ahi[j]; }

    #pragma unroll
    for (int ct = 0; ct < 16; ++ct) {
      // B fragment: col = ct*16 + l%16, K = ks*32 + (l/16)*16 + j
      // Wpack is [col][k] so the 16 K values are 32 contiguous bytes.
      const u16* bp = Wpack + (size_t)(ct * 16 + lmod) * KPAD + ks * 32 + lhi * 16;
      v16bf bfrag = *reinterpret_cast<const v16bf*>(bp);
      acc[ct] = __builtin_amdgcn_wmma_f32_16x16x32_bf16(
          false, a, false, bfrag, (short)0, acc[ct], false, false);
    }
  }

  // Epilogue: C-frag element r holds (M = lhi*8 + r, col = ct*16 + lmod).
  // Gates i/f/o/g live in col-tiles ct, ct+4, ct+8, ct+12 of the SAME lane.
  const int rowbase = rowblock + w * 16 + lhi * 8;
  #pragma unroll
  for (int ct = 0; ct < 4; ++ct) {
    int u = ct * 16 + lmod;
    float bi = bias[u];
    float bf_ = bias[64 + u];
    float bo = bias[128 + u];
    float bg = bias[192 + u];
    #pragma unroll
    for (int r = 0; r < 8; ++r) {
      size_t oi = (size_t)(rowbase + r) * UU + u;   // (b*N+n)*64 + u
      float gi = acc[ct][r]      + bi;
      float gf = acc[ct + 4][r]  + bf_;
      float go = acc[ct + 8][r]  + bo;
      float gg = acc[ct + 12][r] + bg;
      gi = 1.f / (1.f + expf(-gi));
      gf = 1.f / (1.f + expf(-gf));
      go = 1.f / (1.f + expf(-go));
      gg = tanhf(gg);
      float nc = gf * cx[oi] + gi * gg;
      out_h[oi] = go * tanhf(nc);
      out_c[oi] = nc;
    }
  }
}

// ---------------- host launcher ----------------
extern "C" void kernel_launch(void* const* d_in, const int* in_sizes, int n_in,
                              void* d_out, int out_size, void* d_ws, size_t ws_size,
                              hipStream_t stream) {
  const float* inputs = (const float*)d_in[0];
  const float* hx     = (const float*)d_in[1];
  const float* cx     = (const float*)d_in[2];
  const float* vals   = (const float*)d_in[3];
  const int*   rows   = (const int*)d_in[4];
  const int*   cols   = (const int*)d_in[5];
  const float* W      = (const float*)d_in[6];
  const float* bias   = (const float*)d_in[7];

  float* out_h = (float*)d_out;
  float* out_c = out_h + (size_t)MROWS * UU;

  // workspace carve-up (256B aligned)
  char* base = (char*)d_ws;
  size_t off = 0;
  auto carve = [&](size_t bytes) -> char* {
    char* p = base + off;
    off = (off + bytes + 255) & ~(size_t)255;
    return p;
  };
  u16*   x0bf      = (u16*)  carve((size_t)NN * FB * sizeof(u16));   // 42.2 MB
  u16*   x1bf      = (u16*)  carve((size_t)NN * FB * sizeof(u16));   // 42.2 MB
  u16*   Wpack     = (u16*)  carve((size_t)GCOLS * KPAD * sizeof(u16));
  int*   counts    = (int*)  carve((size_t)NN * sizeof(int));
  int*   row_start = (int*)  carve((size_t)(NN + 1) * sizeof(int));
  int*   cursor    = (int*)  carve((size_t)NN * sizeof(int));
  int*   col_s     = (int*)  carve((size_t)EE * sizeof(int));
  float* val_s     = (float*)carve((size_t)EE * sizeof(float));
  (void)ws_size; (void)in_sizes; (void)n_in; (void)out_size;

  // 1) CSR build
  zero_counts_kernel<<<(NN + 255) / 256, 256, 0, stream>>>(counts, NN);
  count_rows_kernel<<<(EE + 255) / 256, 256, 0, stream>>>(rows, counts, EE);
  scan_kernel<<<1, 1024, 0, stream>>>(counts, row_start, cursor, NN);
  scatter_kernel<<<(EE + 255) / 256, 256, 0, stream>>>(rows, cols, vals, cursor,
                                                       col_s, val_s, EE);
  // 2) pack activations and weights to bf16
  pack_x0_kernel<<<(NN * FB + 255) / 256, 256, 0, stream>>>(inputs, hx, x0bf);
  pack_w_kernel<<<(GCOLS * KPAD + 255) / 256, 256, 0, stream>>>(W, Wpack);
  // 3) SpMM x1 = S @ x0 (L2-resident gather)
  spmm_kernel<<<NN, 256, 0, stream>>>(x0bf, row_start, col_s, val_s, x1bf);
  // 4) fused bf16-WMMA GEMM + LSTM gates
  gemm_lstm_kernel<<<MROWS / 128, 256, 0, stream>>>(x0bf, x1bf, Wpack, bias, cx,
                                                    out_h, out_c);
}
